// decoder_52072183497351
// MI455X (gfx1250) — matrix operand order repaired and checked by reading, not compiled, not run on hardware
//
#include <hip/hip_runtime.h>
#include <stdint.h>

// ---------------- types ----------------
typedef __bf16 bf16_t;
typedef __attribute__((ext_vector_type(16))) __bf16 bf16x16;
typedef __attribute__((ext_vector_type(8)))  float  f32x8;
typedef __attribute__((ext_vector_type(4)))  unsigned int u32x4;

// async-load builtin wants int-vector pointers with explicit address spaces
typedef int v4i __attribute__((vector_size(16)));
typedef __attribute__((address_space(1))) v4i* gv4i_p;   // global
typedef __attribute__((address_space(3))) v4i* lv4i_p;   // LDS

#define TB 256

// CDNA5 async global->LDS path (ASYNCcnt), if the toolchain exposes it.
#if __has_builtin(__builtin_amdgcn_global_load_async_to_lds_b128) && \
    __has_builtin(__builtin_amdgcn_s_wait_asynccnt)
#define USE_ASYNC_LDS 1
#else
#define USE_ASYNC_LDS 0
#endif

// ---------------- problem constants ----------------
static constexpr int B_  = 8;
static constexpr int S_  = 2048;
static constexpr int SC_ = 2048;
static constexpr int H_  = 512;
static constexpr int NB_ = 4;
static constexpr int M_  = B_ * S_;      // 16384 token rows
static constexpr float EPS_ = 1e-3f;

// =====================================================================
// GEMM: C[M][N] = scale * (A[M][K] @ Bt[N][K]^T) + bias[N]
//   A  : bf16 row-major [M][K]; Bt : bf16 row-major [N][K]
//   OUTF=true -> fp32 out, OUTF=false -> bf16 out. bias always valid.
// Tiles: BM=128 BN=128 BK=32; 8 wave32 waves (2x4); wave tile 64x32.
// Double-buffered LDS; async global->LDS prefetch when available.
// Requires M%128==0, N%128==0, K%32==0 (true for all uses here).
// =====================================================================
#define BM 128
#define BN 128
#define BK 32
#define LDSP 40   // padded LDS pitch in bf16 elems (80B: 16B-aligned, de-conflicted)

union FragCvt { u32x4 u[2]; bf16x16 v; };

template <bool OUTF>
__global__ __launch_bounds__(TB)
void gemm_bf16_wmma(const bf16_t* __restrict__ A,
                    const bf16_t* __restrict__ Bt,
                    const float*  __restrict__ bias,
                    void* __restrict__ Cout,
                    int M, int N, int K, float scale)
{
    __shared__ __align__(16) bf16_t As[2][BM * LDSP];
    __shared__ __align__(16) bf16_t Bs[2][BN * LDSP];

    const int tid  = threadIdx.x;
    const int lane = tid & 31;
    const int wave = tid >> 5;
    const int wm   = wave >> 2;          // 0..1  (M slab of 64)
    const int wn   = wave & 3;           // 0..3  (N slab of 32)
    const int m0   = blockIdx.y * BM;
    const int n0   = blockIdx.x * BN;

    f32x8 acc[4][2];
    #pragma unroll
    for (int mi = 0; mi < 4; ++mi)
        #pragma unroll
        for (int ni = 0; ni < 2; ++ni)
            #pragma unroll
            for (int e = 0; e < 8; ++e) acc[mi][ni][e] = 0.0f;

    // cooperative tile fetch: 512 chunks of 8 bf16 per tile; 2 chunks/thread
    auto issueTile = [&](int buf, int k0) {
        #pragma unroll
        for (int it = 0; it < 2; ++it) {
            const int c   = tid + it * TB;
            const int row = c >> 2;
            const int col = (c & 3) << 3;
            const bf16_t* ga = &A [(size_t)(m0 + row) * K + k0 + col];
            const bf16_t* gb = &Bt[(size_t)(n0 + row) * K + k0 + col];
            bf16_t* la = &As[buf][row * LDSP + col];
            bf16_t* lb = &Bs[buf][row * LDSP + col];
#if USE_ASYNC_LDS
            __builtin_amdgcn_global_load_async_to_lds_b128(
                (gv4i_p)(unsigned long long)(uintptr_t)ga,
                (lv4i_p)(unsigned)(uintptr_t)la, 0, 0);
            __builtin_amdgcn_global_load_async_to_lds_b128(
                (gv4i_p)(unsigned long long)(uintptr_t)gb,
                (lv4i_p)(unsigned)(uintptr_t)lb, 0, 0);
#else
            *(u32x4*)la = *(const u32x4*)ga;
            *(u32x4*)lb = *(const u32x4*)gb;
#endif
        }
    };

    issueTile(0, 0);
    const int nk = K / BK;
    for (int ik = 0; ik < nk; ++ik) {
        const int cur = ik & 1;
#if USE_ASYNC_LDS
        __builtin_amdgcn_s_wait_asynccnt(0);   // my async writes into 'cur' done
#endif
        __syncthreads();                       // publishes LDS; drains DS reads of 'cur^1'
        if (ik + 1 < nk) issueTile(cur ^ 1, (ik + 1) * BK);

        // Fragment builds per CDNA5 16-bit WMMA VGPR layouts (wave32):
        //  A: lane<16 -> K {0..7,16..23}; lane>=16 -> K {8..15,24..31}
        //  B: lane<16 -> K {0..7,16..23}; lane>=16 -> K {8..15,24..31}
        bf16x16 afrag[4], bfrag[2];
        const int koffA = (lane & 16) ? 8  : 0;
        const int koffB = (lane & 16) ? 8  : 0;
        #pragma unroll
        for (int mi = 0; mi < 4; ++mi) {
            const int r = wm * 64 + mi * 16 + (lane & 15);
            FragCvt f;
            f.u[0] = *(const u32x4*)&As[cur][r * LDSP + koffA];
            f.u[1] = *(const u32x4*)&As[cur][r * LDSP + koffA + 16];
            afrag[mi] = f.v;
        }
        #pragma unroll
        for (int ni = 0; ni < 2; ++ni) {
            const int r = wn * 32 + ni * 16 + (lane & 15);
            FragCvt f;
            f.u[0] = *(const u32x4*)&Bs[cur][r * LDSP + koffB];
            f.u[1] = *(const u32x4*)&Bs[cur][r * LDSP + koffB + 16];
            bfrag[ni] = f.v;
        }
        #pragma unroll
        for (int mi = 0; mi < 4; ++mi)
            #pragma unroll
            for (int ni = 0; ni < 2; ++ni)
                acc[mi][ni] = __builtin_amdgcn_wmma_f32_16x16x32_bf16(
                    false, afrag[mi], false, bfrag[ni],
                    (short)0, acc[mi][ni], false, false);
    }

    // Epilogue: C layout — lane l holds column l%16, rows j + 8*(l>=16)
    float*  __restrict__ cf = (float*)Cout;
    bf16_t* __restrict__ cb = (bf16_t*)Cout;
    const int ncolIn  = lane & 15;
    const int rowHalf = (lane >> 4) * 8;
    #pragma unroll
    for (int mi = 0; mi < 4; ++mi) {
        #pragma unroll
        for (int ni = 0; ni < 2; ++ni) {
            const int ncol = n0 + wn * 32 + ni * 16 + ncolIn;
            const float bv = bias[ncol];
            const size_t rbase =
                (size_t)(m0 + wm * 64 + mi * 16 + rowHalf) * N + ncol;
            #pragma unroll
            for (int j = 0; j < 8; ++j) {
                const float v = acc[mi][ni][j] * scale + bv;
                if (OUTF) cf[rbase + (size_t)j * N] = v;
                else      cb[rbase + (size_t)j * N] = (bf16_t)v;
            }
        }
    }
}

// =====================================================================
// Row softmax over 2048 fp32 logits -> bf16 probabilities. One WG/row.
// =====================================================================
__global__ __launch_bounds__(TB)
void softmax_rows(const float* __restrict__ S, bf16_t* __restrict__ P, int len)
{
    __shared__ float red[TB];
    const size_t row = blockIdx.x;
    const float* s = S + row * (size_t)len;
    bf16_t*      o = P + row * (size_t)len;
    float v[8];
    float mx = -3.0e38f;
    #pragma unroll
    for (int i = 0; i < 8; ++i) { v[i] = s[threadIdx.x + i * TB]; mx = fmaxf(mx, v[i]); }
    red[threadIdx.x] = mx; __syncthreads();
    for (int off = TB / 2; off > 0; off >>= 1) {
        if (threadIdx.x < off) red[threadIdx.x] = fmaxf(red[threadIdx.x], red[threadIdx.x + off]);
        __syncthreads();
    }
    mx = red[0]; __syncthreads();
    float sum = 0.0f;
    #pragma unroll
    for (int i = 0; i < 8; ++i) { v[i] = __expf(v[i] - mx); sum += v[i]; }
    red[threadIdx.x] = sum; __syncthreads();
    for (int off = TB / 2; off > 0; off >>= 1) {
        if (threadIdx.x < off) red[threadIdx.x] += red[threadIdx.x + off];
        __syncthreads();
    }
    const float inv = 1.0f / red[0];
    #pragma unroll
    for (int i = 0; i < 8; ++i) o[threadIdx.x + i * TB] = (bf16_t)(v[i] * inv);
}

// =====================================================================
// Fused residual-add + LayerNorm over H=512. One WG per token row.
// =====================================================================
__global__ __launch_bounds__(TB)
void add_layernorm(const float* __restrict__ X, const float* __restrict__ Y,
                   const float* __restrict__ gamma, const float* __restrict__ beta,
                   float* __restrict__ outF, bf16_t* __restrict__ outB)
{
    __shared__ float r1[TB], r2[TB];
    const size_t base = blockIdx.x * (size_t)H_;
    float z[2], sum = 0.0f, sum2 = 0.0f;
    #pragma unroll
    for (int i = 0; i < 2; ++i) {
        const int c = threadIdx.x + i * TB;
        const float v = X[base + c] + Y[base + c];
        z[i] = v; sum += v; sum2 += v * v;
    }
    r1[threadIdx.x] = sum; r2[threadIdx.x] = sum2; __syncthreads();
    for (int off = TB / 2; off > 0; off >>= 1) {
        if (threadIdx.x < off) {
            r1[threadIdx.x] += r1[threadIdx.x + off];
            r2[threadIdx.x] += r2[threadIdx.x + off];
        }
        __syncthreads();
    }
    const float mean = r1[0] * (1.0f / H_);
    const float var  = r2[0] * (1.0f / H_) - mean * mean;
    const float inv  = rsqrtf(var + EPS_);
    #pragma unroll
    for (int i = 0; i < 2; ++i) {
        const int c = threadIdx.x + i * TB;
        const float o = gamma[c] * (z[i] - mean) * inv + beta[c];
        outF[base + c] = o;
        if (outB) outB[base + c] = (bf16_t)o;
    }
}

// =====================================================================
// Embedding gather + sin/cos positional encoding (base 1000, concat).
// =====================================================================
__global__ __launch_bounds__(TB)
void embed_pos(const int* __restrict__ x, const float* __restrict__ emb,
               float* __restrict__ h, bf16_t* __restrict__ hb)
{
    const size_t row = blockIdx.x;            // b*S + s
    const int s   = (int)(row % S_);
    const int tok = x[row];
    const float* e = emb + (size_t)tok * H_;
    #pragma unroll
    for (int i = 0; i < 2; ++i) {
        const int c = threadIdx.x + i * TB;
        const int half = H_ >> 1;
        const int j = (c < half) ? c : c - half;
        const float freq = __expf(-(2.0f * (float)j / (float)H_) * 6.907755278982137f);
        const float a = (float)s * freq;
        const float pv = (c < half) ? __sinf(a) : __cosf(a);
        const float v = e[c] + pv;
        h [row * (size_t)H_ + c] = v;
        hb[row * (size_t)H_ + c] = (bf16_t)v;
    }
}

// =====================================================================
// Tiled transposes (32x32 tiles, LDS staging). Batched over blockIdx.z.
// =====================================================================
__global__ __launch_bounds__(TB)
void transpose_f32_bf16(const float* __restrict__ in, bf16_t* __restrict__ out,
                        int R, int C)
{
    __shared__ float tile[32][33];
    const size_t zo = (size_t)blockIdx.z * (size_t)R * (size_t)C;
    const int c0 = blockIdx.x * 32, r0 = blockIdx.y * 32;
    const int tx = threadIdx.x & 31, ty = threadIdx.x >> 5;
    #pragma unroll
    for (int i = 0; i < 4; ++i) {
        const int r = ty + i * 8;
        tile[r][tx] = in[zo + (size_t)(r0 + r) * C + c0 + tx];
    }
    __syncthreads();
    #pragma unroll
    for (int i = 0; i < 4; ++i) {
        const int r = ty + i * 8;
        out[zo + (size_t)(c0 + r) * R + r0 + tx] = (bf16_t)tile[tx][r];
    }
}

__global__ __launch_bounds__(TB)
void transpose_bf16(const bf16_t* __restrict__ in, bf16_t* __restrict__ out,
                    int R, int C)
{
    __shared__ bf16_t tile[32][34];
    const size_t zo = (size_t)blockIdx.z * (size_t)R * (size_t)C;
    const int c0 = blockIdx.x * 32, r0 = blockIdx.y * 32;
    const int tx = threadIdx.x & 31, ty = threadIdx.x >> 5;
    #pragma unroll
    for (int i = 0; i < 4; ++i) {
        const int r = ty + i * 8;
        tile[r][tx] = in[zo + (size_t)(r0 + r) * C + c0 + tx];
    }
    __syncthreads();
    #pragma unroll
    for (int i = 0; i < 4; ++i) {
        const int r = ty + i * 8;
        out[zo + (size_t)(c0 + r) * R + r0 + tx] = tile[tx][r];
    }
}

__global__ __launch_bounds__(TB)
void cvt_f32_bf16(const float* __restrict__ in, bf16_t* __restrict__ out, long long n)
{
    long long i = (long long)blockIdx.x * TB + threadIdx.x;
    const long long st = (long long)gridDim.x * TB;
    for (; i < n; i += st) out[i] = (bf16_t)in[i];
}

__global__ __launch_bounds__(TB)
void fill_zero_f32(float* __restrict__ p, int n)
{
    for (int i = threadIdx.x + blockIdx.x * TB; i < n; i += gridDim.x * TB)
        p[i] = 0.0f;
}

// =====================================================================
// Host orchestration
// =====================================================================
extern "C" void kernel_launch(void* const* d_in, const int* in_sizes, int n_in,
                              void* d_out, int out_size, void* d_ws, size_t ws_size,
                              hipStream_t stream)
{
    (void)in_sizes; (void)n_in; (void)out_size; (void)ws_size;

    const float* ctx   = (const float*)d_in[0];
    const int*   x     = (const int*)  d_in[1];
    const float* emb   = (const float*)d_in[2];
    const float* Wq_s  = (const float*)d_in[3];
    const float* bq_s  = (const float*)d_in[4];
    const float* Wk_s  = (const float*)d_in[5];
    const float* bk_s  = (const float*)d_in[6];
    const float* Wv_s  = (const float*)d_in[7];
    const float* bv_s  = (const float*)d_in[8];
    const float* Wq_c  = (const float*)d_in[9];
    const float* bq_c  = (const float*)d_in[10];
    const float* Wk_c  = (const float*)d_in[11];
    const float* bk_c  = (const float*)d_in[12];
    const float* Wv_c  = (const float*)d_in[13];
    const float* bv_c  = (const float*)d_in[14];
    const float* Wd    = (const float*)d_in[15];
    const float* bd    = (const float*)d_in[16];
    const float* ga1   = (const float*)d_in[17];
    const float* be1   = (const float*)d_in[18];
    const float* ga2   = (const float*)d_in[19];
    const float* be2   = (const float*)d_in[20];
    const float* ga3   = (const float*)d_in[21];
    const float* be3   = (const float*)d_in[22];

    // ---- deterministic workspace carve ----
    char* p = (char*)d_ws;
    auto take = [&](size_t bytes) -> char* {
        char* r = p; p += (bytes + 255) & ~(size_t)255; return r;
    };
    const size_t MH = (size_t)M_ * H_;
    const size_t SS = (size_t)S_ * S_;
    const size_t HH = (size_t)H_ * H_;

    float*  hf0  = (float*) take(MH * 4);
    float*  hf1  = (float*) take(MH * 4);
    float*  hf2  = (float*) take(MH * 4);
    float*  tmpf = (float*) take(MH * 4);
    bf16_t* hb0  = (bf16_t*)take(MH * 2);
    bf16_t* hb1  = (bf16_t*)take(MH * 2);
    bf16_t* hb2  = (bf16_t*)take(MH * 2);
    bf16_t* qb   = (bf16_t*)take(MH * 2);
    bf16_t* kb   = (bf16_t*)take(MH * 2);
    bf16_t* vb   = (bf16_t*)take(MH * 2);
    bf16_t* vtb  = (bf16_t*)take(MH * 2);
    bf16_t* ctxb = (bf16_t*)take(MH * 2);
    bf16_t* wt   = (bf16_t*)take((size_t)(6 * NB_ + 1) * HH * 2);
    float*  scr  = (float*) take(SS * 4);
    bf16_t* prb  = (bf16_t*)take(SS * 2);
    float*  zb   = (float*) take((size_t)H_ * 4);   // zero bias for attention GEMMs

    bf16_t* wqs_t = wt;
    bf16_t* wks_t = wqs_t + (size_t)NB_ * HH;
    bf16_t* wvs_t = wks_t + (size_t)NB_ * HH;
    bf16_t* wqc_t = wvs_t + (size_t)NB_ * HH;
    bf16_t* wkc_t = wqc_t + (size_t)NB_ * HH;
    bf16_t* wvc_t = wkc_t + (size_t)NB_ * HH;
    bf16_t* wd_t  = wvc_t + (size_t)NB_ * HH;

    const dim3 blk(TB);

    // ---- prep ----
    {
        const dim3 gw(H_ / 32, H_ / 32, NB_);
        transpose_f32_bf16<<<gw, blk, 0, stream>>>(Wq_s, wqs_t, H_, H_);
        transpose_f32_bf16<<<gw, blk, 0, stream>>>(Wk_s, wks_t, H_, H_);
        transpose_f32_bf16<<<gw, blk, 0, stream>>>(Wv_s, wvs_t, H_, H_);
        transpose_f32_bf16<<<gw, blk, 0, stream>>>(Wq_c, wqc_t, H_, H_);
        transpose_f32_bf16<<<gw, blk, 0, stream>>>(Wk_c, wkc_t, H_, H_);
        transpose_f32_bf16<<<gw, blk, 0, stream>>>(Wv_c, wvc_t, H_, H_);
        const dim3 gw1(H_ / 32, H_ / 32, 1);
        transpose_f32_bf16<<<gw1, blk, 0, stream>>>(Wd, wd_t, H_, H_);
    }
    fill_zero_f32<<<dim3(1), blk, 0, stream>>>(zb, H_);
    cvt_f32_bf16<<<dim3(4096), blk, 0, stream>>>(ctx, ctxb, (long long)MH);
    embed_pos<<<dim3(M_), blk, 0, stream>>>(x, emb, hf0, hb0);

    const float attn_scale = 0.044194173824159216f; // 1/sqrt(512)

    auto gemmF = [&](const bf16_t* Aq, const bf16_t* Bq, const float* bias,
                     float* Cf, int Mm, int Nn, int Kk, float sc) {
        const dim3 grid(Nn / BN, Mm / BM, 1);
        gemm_bf16_wmma<true><<<grid, blk, 0, stream>>>(Aq, Bq, bias, (void*)Cf,
                                                       Mm, Nn, Kk, sc);
    };
    auto gemmB = [&](const bf16_t* Aq, const bf16_t* Bq, const float* bias,
                     bf16_t* Cb, int Mm, int Nn, int Kk, float sc) {
        const dim3 grid(Nn / BN, Mm / BM, 1);
        gemm_bf16_wmma<false><<<grid, blk, 0, stream>>>(Aq, Bq, bias, (void*)Cb,
                                                        Mm, Nn, Kk, sc);
    };

    for (int i = 0; i < NB_; ++i) {
        const size_t wOff = (size_t)i * HH;
        const size_t bOff = (size_t)i * H_;

        // ---------- decoder self-attention ----------
        gemmB(hb0, wqs_t + wOff, bq_s + bOff, qb, M_, H_, H_, 1.0f);
        gemmB(hb0, wks_t + wOff, bk_s + bOff, kb, M_, H_, H_, 1.0f);
        gemmB(hb0, wvs_t + wOff, bv_s + bOff, vb, M_, H_, H_, 1.0f);
        transpose_bf16<<<dim3(H_ / 32, S_ / 32, B_), blk, 0, stream>>>(vb, vtb, S_, H_);
        for (int b = 0; b < B_; ++b) {
            const size_t o = (size_t)b * S_ * H_;
            gemmF(qb + o, kb + o, zb, scr, S_, S_, H_, attn_scale);
            softmax_rows<<<dim3(S_), blk, 0, stream>>>(scr, prb, S_);
            gemmF(prb, vtb + o, zb, tmpf + o, S_, H_, S_, 1.0f);
        }
        add_layernorm<<<dim3(M_), blk, 0, stream>>>(hf0, tmpf, ga1, be1, hf1, hb1);

        // ---------- cross-attention (unscaled logits, per reference) ----------
        gemmB(hb1,  wqc_t + wOff, bq_c + bOff, qb, M_, H_, H_, 1.0f);
        gemmB(ctxb, wkc_t + wOff, bk_c + bOff, kb, M_, H_, H_, 1.0f);
        gemmB(ctxb, wvc_t + wOff, bv_c + bOff, vb, M_, H_, H_, 1.0f);
        transpose_bf16<<<dim3(H_ / 32, SC_ / 32, B_), blk, 0, stream>>>(vb, vtb, SC_, H_);
        for (int b = 0; b < B_; ++b) {
            const size_t oq = (size_t)b * S_  * H_;
            const size_t ok = (size_t)b * SC_ * H_;
            gemmF(qb + oq, kb + ok, zb, scr, S_, SC_, H_, 1.0f);
            softmax_rows<<<dim3(S_), blk, 0, stream>>>(scr, prb, SC_);
            gemmF(prb, vtb + ok, zb, tmpf + oq, S_, H_, SC_, 1.0f);
        }
        add_layernorm<<<dim3(M_), blk, 0, stream>>>(hf1, tmpf, ga2, be2, hf2, hb2);

        // ---------- shared dense FFN (no activation) ----------
        gemmF(hb2, wd_t, bd, tmpf, M_, H_, H_, 1.0f);
        const bool last = (i == NB_ - 1);
        add_layernorm<<<dim3(M_), blk, 0, stream>>>(
            hf2, tmpf, ga3, be3,
            last ? (float*)d_out : hf0,
            last ? (bf16_t*)nullptr : hb0);
    }
}
